// FlashAttention_85950885527669
// MI455X (gfx1250) — compile-verified, matrix operand order checked by reading, not run on hardware
//
#include <hip/hip_runtime.h>

#define T_SEQ      4096
#define HIDDEN     2048
#define N_HEADS    16
#define N_KV_HEADS 4
#define HEAD_DIM   128
#define WINDOW     1024
#define ROPE_THETA 100000.0f
#define QKV_OUT    ((N_HEADS + 2 * N_KV_HEADS) * HEAD_DIM)   // 3072
#define Q_SIZE     (N_HEADS * HEAD_DIM)                       // 2048
#define KV_SIZE    (N_KV_HEADS * HEAD_DIM)                    // 512

typedef __attribute__((ext_vector_type(16))) _Float16 v16h;
typedef __attribute__((ext_vector_type(8)))  _Float16 v8h;
typedef __attribute__((ext_vector_type(8)))  float    v8f;

union H16 { v16h v; v8h h[2]; };

// CDNA5 async global->LDS 16-byte copy (ASYNCcnt-tracked, bypasses VGPRs).
// Low 32 bits of a generic pointer to __shared__ are the LDS byte offset.
__device__ __forceinline__ void async_copy16(void* ldsPtr, const void* gPtr) {
    unsigned lds_off = (unsigned)(unsigned long long)(uintptr_t)ldsPtr;
    unsigned long long ga = (unsigned long long)(uintptr_t)gPtr;
    asm volatile("global_load_async_to_lds_b128 %0, %1, off"
                 :: "v"(lds_off), "v"(ga) : "memory");
}
__device__ __forceinline__ void wait_async0() {
    asm volatile("s_wait_asynccnt 0x0" ::: "memory");
}

__device__ __forceinline__ v8h cvt8(const float* __restrict__ p) {
    float4 x = *(const float4*)p;
    float4 y = *(const float4*)(p + 4);
    v8h r;
    r[0] = (_Float16)x.x; r[1] = (_Float16)x.y; r[2] = (_Float16)x.z; r[3] = (_Float16)x.w;
    r[4] = (_Float16)y.x; r[5] = (_Float16)y.y; r[6] = (_Float16)y.z; r[7] = (_Float16)y.w;
    return r;
}

// ---------------------------------------------------------------------------
// f32 -> f16 elementwise convert (for hidden_states).
// ---------------------------------------------------------------------------
__global__ __launch_bounds__(256)
void cvt_f16_kernel(const float* __restrict__ src, _Float16* __restrict__ dst, int n) {
    int i = (blockIdx.x * 256 + threadIdx.x) * 4;
    if (i < n) {
        float4 f = *(const float4*)(src + i);
        dst[i + 0] = (_Float16)f.x;
        dst[i + 1] = (_Float16)f.y;
        dst[i + 2] = (_Float16)f.z;
        dst[i + 3] = (_Float16)f.w;
    }
}

// ---------------------------------------------------------------------------
// Transpose + convert: src f32 [K][N] -> dst f16 [N][K]. 32x32 LDS tiles.
// ---------------------------------------------------------------------------
__global__ __launch_bounds__(256)
void transpose_cvt_kernel(const float* __restrict__ src, _Float16* __restrict__ dst,
                          int K, int N) {
    __shared__ float tile[32][33];
    const int bn = blockIdx.x * 32;
    const int bk = blockIdx.y * 32;
    const int tx = threadIdx.x & 31;
    const int ty = threadIdx.x >> 5;           // 0..7
#pragma unroll
    for (int j = 0; j < 32; j += 8)
        tile[ty + j][tx] = src[(size_t)(bk + ty + j) * N + bn + tx];
    __syncthreads();
#pragma unroll
    for (int j = 0; j < 32; j += 8)
        dst[(size_t)(bn + ty + j) * K + bk + tx] = (_Float16)tile[tx][ty + j];
}

// ---------------------------------------------------------------------------
// GEMM: C[M,N](f32,+bias) = A[M,K](f16 row-major) @ BT[N,K](f16, B transposed).
// Block = 256 threads (8 waves). Tile 128x128, K-chunks of 32, double-buffered
// LDS fed by async global->LDS copies. Each wave: 32(M) x 64(N) = 8 WMMA/chunk.
// ---------------------------------------------------------------------------
#define BM 128
#define BN 128
#define BK 32

__global__ __launch_bounds__(256)
void gemm_f16_async(const _Float16* __restrict__ A, const _Float16* __restrict__ BT,
                    const float* __restrict__ bias, float* __restrict__ C,
                    int M, int N, int K) {
    __shared__ __align__(16) _Float16 As[2][BM][BK + 8];   // [m][k], pitch 80B
    __shared__ __align__(16) _Float16 Bs[2][BN][BK + 8];   // [n][k], pitch 80B

    const int tid  = threadIdx.x;
    const int wv   = tid >> 5;
    const int lane = tid & 31;
    const int hx   = lane >> 4;
    const int l16  = lane & 15;
    const int K0   = hx ? 8 : 0;

    const int blockN = blockIdx.x * BN;
    const int blockM = blockIdx.y * BM;
    const int M0 = (wv >> 1) * 32;             // wave's M offset within tile
    const int N0 = (wv & 1) * 64;              // wave's N offset within tile

    v8f acc[2][4] = {};

    // Stage one K-chunk: 512 16B chunks for A, 512 for B; 2+2 per thread.
    auto stage = [&](int b, int kk) {
#pragma unroll
        for (int i = 0; i < 2; ++i) {
            int idx = tid + 256 * i;
            int r   = idx >> 2;                // 4 chunks (64B) per row
            int c   = idx & 3;
            async_copy16(&As[b][r][c * 8],
                         A + (size_t)(blockM + r) * K + kk + c * 8);
        }
#pragma unroll
        for (int i = 0; i < 2; ++i) {
            int idx = tid + 256 * i;
            int r   = idx >> 2;
            int c   = idx & 3;
            async_copy16(&Bs[b][r][c * 8],
                         BT + (size_t)(blockN + r) * K + kk + c * 8);
        }
    };

    stage(0, 0);
    int buf = 0;
    for (int k0 = 0; k0 < K; k0 += BK) {
        wait_async0();                          // my async copies to buf done
        __syncthreads();                        // everyone's copies done
        if (k0 + BK < K) stage(buf ^ 1, k0 + BK);   // overlap next tile w/ compute

        // Load all fragments, then burst 8 WMMAs.
        H16 a[2], b[4];
#pragma unroll
        for (int mt = 0; mt < 2; ++mt) {
            a[mt].h[0] = *(const v8h*)&As[buf][M0 + mt * 16 + l16][K0];
            a[mt].h[1] = *(const v8h*)&As[buf][M0 + mt * 16 + l16][K0 + 16];
        }
#pragma unroll
        for (int nt = 0; nt < 4; ++nt) {
            b[nt].h[0] = *(const v8h*)&Bs[buf][N0 + nt * 16 + l16][K0];
            b[nt].h[1] = *(const v8h*)&Bs[buf][N0 + nt * 16 + l16][K0 + 16];
        }
#pragma unroll
        for (int mt = 0; mt < 2; ++mt)
#pragma unroll
            for (int nt = 0; nt < 4; ++nt)
                acc[mt][nt] = __builtin_amdgcn_wmma_f32_16x16x32_f16(
                    false, a[mt].v, false, b[nt].v, (short)0, acc[mt][nt], false, false);

        buf ^= 1;
        // Buffer-reuse hazard is covered by next iteration's barrier.
    }

    // C-layout scatter: VGPR r, lane -> (M = r + 8*hx, N = l16).
#pragma unroll
    for (int mt = 0; mt < 2; ++mt)
#pragma unroll
        for (int nt = 0; nt < 4; ++nt) {
            int n = blockN + N0 + nt * 16 + l16;
            float bv = bias ? bias[n] : 0.0f;
#pragma unroll
            for (int r = 0; r < 8; ++r) {
                int m = blockM + M0 + mt * 16 + r + hx * 8;
                C[(size_t)m * N + n] = acc[mt][nt][r] + bv;
            }
        }
}

// ---------------------------------------------------------------------------
// RoPE: in-place on f32 qkv workspace (16 q-heads + 4 k-heads).
// ---------------------------------------------------------------------------
__global__ __launch_bounds__(64)
void rope_kernel(float* __restrict__ qkv, const int* __restrict__ positions) {
    const int t  = blockIdx.x;
    const int hh = blockIdx.y;
    const int i  = threadIdx.x;
    size_t base = (size_t)t * QKV_OUT +
                  (hh < N_HEADS ? hh * HEAD_DIM
                                : Q_SIZE + (hh - N_HEADS) * HEAD_DIM);
    float inv = __powf(ROPE_THETA, -(float)i / 64.0f);
    float f   = (float)positions[t] * inv;
    float c   = __cosf(f);
    float s   = __sinf(f);
    float x1 = qkv[base + i];
    float x2 = qkv[base + 64 + i];
    qkv[base + i]      = x1 * c - x2 * s;
    qkv[base + 64 + i] = x2 * c + x1 * s;
}

// ---------------------------------------------------------------------------
// Sliding-window flash attention with WMMA. Output written as f16.
// grid = (T/64, N_HEADS), block = 128 (4 waves, 16 queries each).
// ---------------------------------------------------------------------------
#define TQ  64
#define TKV 32

__global__ __launch_bounds__(128)
void attn_wmma(const float* __restrict__ qkv, _Float16* __restrict__ attn_out) {
    __shared__ __align__(16) _Float16 Ks[TKV][HEAD_DIM + 8];   // [k][d]
    __shared__ __align__(16) _Float16 Vs[HEAD_DIM][TKV + 8];   // [d][k]
    __shared__ __align__(16) _Float16 Ps[4][16][TKV + 8];      // per-wave P tile

    const int tid  = threadIdx.x;
    const int wv   = tid >> 5;
    const int lane = tid & 31;
    const int hx   = lane >> 4;
    const int l16  = lane & 15;
    const int K0   = hx ? 8 : 0;

    const int h   = blockIdx.y;
    const int kvh = h >> 2;
    const int qb  = blockIdx.x;
    const int q0  = qb * TQ + wv * 16;

    // Q fragments (4 d-chunks) in registers, f32 -> f16 on load.
    H16 qf[4];
    {
        const float* qrow = qkv + (size_t)(q0 + l16) * QKV_OUT + h * HEAD_DIM;
#pragma unroll
        for (int dc = 0; dc < 4; ++dc) {
            qf[dc].h[0] = cvt8(qrow + dc * 32 + K0);
            qf[dc].h[1] = cvt8(qrow + dc * 32 + K0 + 16);
        }
    }

    float m_row[8], l_row[8];
#pragma unroll
    for (int r = 0; r < 8; ++r) { m_row[r] = -3.0e38f; l_row[r] = 0.0f; }
    v8f accO[8] = {};

    const float scale = 0.08838834764831845f;

    int kcBeg = qb * TQ - WINDOW;
    if (kcBeg < 0) kcBeg = 0;
    const int kcEnd = qb * TQ + TQ;

    for (int kc = kcBeg; kc < kcEnd; kc += TKV) {
        __syncthreads();
        // Batched staging: issue all 16 global loads, then convert + store.
        float4 kr[8], vr[8];
#pragma unroll
        for (int i = 0; i < 8; ++i) {
            int idx = tid + 128 * i;
            int r   = idx >> 5;
            int c4  = idx & 31;
            kr[i] = *(const float4*)(qkv + (size_t)(kc + r) * QKV_OUT + Q_SIZE +
                                     kvh * HEAD_DIM + c4 * 4);
            vr[i] = *(const float4*)(qkv + (size_t)(kc + r) * QKV_OUT + Q_SIZE + KV_SIZE +
                                     kvh * HEAD_DIM + c4 * 4);
        }
#pragma unroll
        for (int i = 0; i < 8; ++i) {
            int idx = tid + 128 * i;
            int r   = idx >> 5;
            int c4  = idx & 31;
            Ks[r][c4 * 4 + 0] = (_Float16)kr[i].x;
            Ks[r][c4 * 4 + 1] = (_Float16)kr[i].y;
            Ks[r][c4 * 4 + 2] = (_Float16)kr[i].z;
            Ks[r][c4 * 4 + 3] = (_Float16)kr[i].w;
            Vs[c4 * 4 + 0][r] = (_Float16)vr[i].x;
            Vs[c4 * 4 + 1][r] = (_Float16)vr[i].y;
            Vs[c4 * 4 + 2][r] = (_Float16)vr[i].z;
            Vs[c4 * 4 + 3][r] = (_Float16)vr[i].w;
        }
        __syncthreads();

        // S = Q.K^T : grouped fragment loads, 4-WMMA bursts.
        v8f s0 = {}, s1 = {};
#pragma unroll
        for (int dcp = 0; dcp < 4; dcp += 2) {
            H16 b[4];
            b[0].h[0] = *(const v8h*)&Ks[l16][dcp * 32 + K0];
            b[0].h[1] = *(const v8h*)&Ks[l16][dcp * 32 + K0 + 16];
            b[1].h[0] = *(const v8h*)&Ks[16 + l16][dcp * 32 + K0];
            b[1].h[1] = *(const v8h*)&Ks[16 + l16][dcp * 32 + K0 + 16];
            b[2].h[0] = *(const v8h*)&Ks[l16][(dcp + 1) * 32 + K0];
            b[2].h[1] = *(const v8h*)&Ks[l16][(dcp + 1) * 32 + K0 + 16];
            b[3].h[0] = *(const v8h*)&Ks[16 + l16][(dcp + 1) * 32 + K0];
            b[3].h[1] = *(const v8h*)&Ks[16 + l16][(dcp + 1) * 32 + K0 + 16];
            s0 = __builtin_amdgcn_wmma_f32_16x16x32_f16(
                false, qf[dcp].v, false, b[0].v, (short)0, s0, false, false);
            s1 = __builtin_amdgcn_wmma_f32_16x16x32_f16(
                false, qf[dcp].v, false, b[1].v, (short)0, s1, false, false);
            s0 = __builtin_amdgcn_wmma_f32_16x16x32_f16(
                false, qf[dcp + 1].v, false, b[2].v, (short)0, s0, false, false);
            s1 = __builtin_amdgcn_wmma_f32_16x16x32_f16(
                false, qf[dcp + 1].v, false, b[3].v, (short)0, s1, false, false);
        }

        // Online softmax (row M = r + 8*hx lives across a 16-lane half).
        float alpha[8];
#pragma unroll
        for (int r = 0; r < 8; ++r) {
            const int qi = q0 + r + hx * 8;
            const int j0 = kc + l16;
            const int j1 = kc + 16 + l16;
            const bool ok0 = (j0 <= qi) && ((qi - j0) <= WINDOW);
            const bool ok1 = (j1 <= qi) && ((qi - j1) <= WINDOW);
            float v0 = ok0 ? s0[r] * scale : -3.0e38f;
            float v1 = ok1 ? s1[r] * scale : -3.0e38f;
            float mx = fmaxf(v0, v1);
#pragma unroll
            for (int off = 1; off < 16; off <<= 1)
                mx = fmaxf(mx, __shfl_xor(mx, off, 32));
            float mnew = fmaxf(m_row[r], mx);
            float p0 = ok0 ? __expf(v0 - mnew) : 0.0f;
            float p1 = ok1 ? __expf(v1 - mnew) : 0.0f;
            float sm = p0 + p1;
#pragma unroll
            for (int off = 1; off < 16; off <<= 1)
                sm += __shfl_xor(sm, off, 32);
            alpha[r] = (m_row[r] > -1.0e37f) ? __expf(m_row[r] - mnew) : 0.0f;
            l_row[r] = l_row[r] * alpha[r] + sm;
            m_row[r] = mnew;
            Ps[wv][r + hx * 8][l16]      = (_Float16)p0;
            Ps[wv][r + hx * 8][16 + l16] = (_Float16)p1;
        }

#pragma unroll
        for (int dt = 0; dt < 8; ++dt)
#pragma unroll
            for (int r = 0; r < 8; ++r)
                accO[dt][r] *= alpha[r];

        // Same-wave cross-lane LDS RAW: drain DS counter explicitly.
        asm volatile("s_wait_dscnt 0x0" ::: "memory");

        // O += P.V : grouped fragment loads, 4-WMMA bursts.
        H16 pf;
        pf.h[0] = *(const v8h*)&Ps[wv][l16][K0];
        pf.h[1] = *(const v8h*)&Ps[wv][l16][K0 + 16];
#pragma unroll
        for (int g = 0; g < 2; ++g) {
            H16 vb[4];
#pragma unroll
            for (int t = 0; t < 4; ++t) {
                int dt = g * 4 + t;
                vb[t].h[0] = *(const v8h*)&Vs[dt * 16 + l16][K0];
                vb[t].h[1] = *(const v8h*)&Vs[dt * 16 + l16][K0 + 16];
            }
#pragma unroll
            for (int t = 0; t < 4; ++t)
                accO[g * 4 + t] = __builtin_amdgcn_wmma_f32_16x16x32_f16(
                    false, pf.v, false, vb[t].v, (short)0, accO[g * 4 + t], false, false);
        }
    }

    // Normalize and write attn (f16) for the projection GEMM.
#pragma unroll
    for (int dt = 0; dt < 8; ++dt) {
        int d = dt * 16 + l16;
#pragma unroll
        for (int r = 0; r < 8; ++r) {
            int q = q0 + r + hx * 8;
            float l = l_row[r];
            float val = (l > 0.0f) ? accO[dt][r] / l : 0.0f;
            attn_out[(size_t)q * Q_SIZE + h * HEAD_DIM + d] = (_Float16)val;
        }
    }
}

// ---------------------------------------------------------------------------
// Launch: converts -> QKV GEMM -> RoPE -> attention -> output GEMM.
// ws layout: qkv f32 (48MB) | h16 (16MB) | wq16T (12MB) | wo16T (8MB) | attn16 (16MB)
// ---------------------------------------------------------------------------
extern "C" void kernel_launch(void* const* d_in, const int* in_sizes, int n_in,
                              void* d_out, int out_size, void* d_ws, size_t ws_size,
                              hipStream_t stream) {
    (void)in_sizes; (void)n_in; (void)out_size; (void)ws_size;
    const float* hidden    = (const float*)d_in[0];
    const int*   positions = (const int*)d_in[1];
    const float* w_qkv     = (const float*)d_in[2];
    const float* b_qkv     = (const float*)d_in[3];
    const float* w_o       = (const float*)d_in[4];
    float* out = (float*)d_out;

    float*    qkv    = (float*)d_ws;
    _Float16* h16    = (_Float16*)((char*)d_ws + (size_t)T_SEQ * QKV_OUT * 4);
    _Float16* wq16T  = h16   + (size_t)T_SEQ * HIDDEN;       // [QKV_OUT][HIDDEN]
    _Float16* wo16T  = wq16T + (size_t)HIDDEN * QKV_OUT;     // [HIDDEN][Q_SIZE]
    _Float16* attn16 = wo16T + (size_t)Q_SIZE * HIDDEN;      // [T][Q_SIZE]

    // Precision pre-pass: f16 activations + transposed f16 weights.
    cvt_f16_kernel<<<(T_SEQ * HIDDEN / 4 + 255) / 256, 256, 0, stream>>>(
        hidden, h16, T_SEQ * HIDDEN);
    transpose_cvt_kernel<<<dim3(QKV_OUT / 32, HIDDEN / 32), 256, 0, stream>>>(
        w_qkv, wq16T, HIDDEN, QKV_OUT);
    transpose_cvt_kernel<<<dim3(HIDDEN / 32, Q_SIZE / 32), 256, 0, stream>>>(
        w_o, wo16T, Q_SIZE, HIDDEN);

    dim3 g1(QKV_OUT / BN, T_SEQ / BM);     // (24, 32)
    gemm_f16_async<<<g1, 256, 0, stream>>>(h16, wq16T, b_qkv, qkv,
                                           T_SEQ, QKV_OUT, HIDDEN);

    dim3 g2(T_SEQ, N_HEADS + N_KV_HEADS);
    rope_kernel<<<g2, 64, 0, stream>>>(qkv, positions);

    dim3 g3(T_SEQ / TQ, N_HEADS);          // (64, 16)
    attn_wmma<<<g3, 128, 0, stream>>>(qkv, attn16);

    dim3 g4(HIDDEN / BN, T_SEQ / BM);      // (16, 32)
    gemm_f16_async<<<g4, 256, 0, stream>>>(attn16, wo16T, nullptr, out,
                                           T_SEQ, HIDDEN, Q_SIZE);
}
